// ds_conv_68332929679860
// MI455X (gfx1250) — matrix operand order repaired and checked by reading, not compiled, not run on hardware
//
#include <hip/hip_runtime.h>
#include <hip/hip_bf16.h>
#include <math.h>
#include <stdint.h>

// ---------------------------------------------------------------------------
// Sizes from the reference
// ---------------------------------------------------------------------------
#define HIN   1024
#define WIN   1024
#define CIN   64
#define COUT  128
#define HD    512          // downsampled H
#define WD    512          // downsampled W
#define NGRP  32
#define GEPS  1e-5f

// ---------------------------------------------------------------------------
// Vector types for WMMA fragments (wave32, gfx1250)
// ---------------------------------------------------------------------------
typedef __attribute__((ext_vector_type(16))) __bf16 v16bf;
typedef __attribute__((ext_vector_type(8)))  __bf16 v8bf;
typedef __attribute__((ext_vector_type(4)))  __bf16 v4bf;
typedef __attribute__((ext_vector_type(8)))  float  v8f;

// GCC-style 16-byte int vector: the payload type the async-LDS builtin wants.
typedef int v4i_vs __attribute__((vector_size(16)));

// ---------------------------------------------------------------------------
// Async global->LDS helpers (gfx1250 GLOBAL_LOAD_ASYNC_TO_LDS_B128, ASYNCcnt).
// Probe result (round 2 stderr): param 0 is
//   '__attribute__((vector_size(16))) int __device__ *'   (addrspace(1) int4*)
// so we cast to the exact vector-pointer types.  Fallback remains the plain
// vector-load + ds_store path.
// ---------------------------------------------------------------------------
#if defined(__gfx1250__) && __has_builtin(__builtin_amdgcn_global_load_async_to_lds_b128)
#define USE_ASYNC_LDS 1
#else
#define USE_ASYNC_LDS 0
#endif

__device__ __forceinline__ void g2l_16B(const void* gsrc, void* ldst) {
#if USE_ASYNC_LDS
    __builtin_amdgcn_global_load_async_to_lds_b128(
        (__attribute__((address_space(1))) v4i_vs*)(uintptr_t)gsrc,
        (__attribute__((address_space(3))) v4i_vs*)(uint32_t)(uintptr_t)ldst,
        /*imm offset=*/0, /*cpol=*/0);
#else
    *(v8bf*)ldst = *(const v8bf*)gsrc;
#endif
}

__device__ __forceinline__ void wait_async0() {
#if USE_ASYNC_LDS
#if __has_builtin(__builtin_amdgcn_s_wait_asynccnt)
    __builtin_amdgcn_s_wait_asynccnt(0);
#else
    asm volatile("s_wait_asynccnt 0" ::: "memory");
#endif
#endif
}

// ---------------------------------------------------------------------------
// Kernel 0: zero the stats accumulators (ws is poisoned by the harness)
// ---------------------------------------------------------------------------
__global__ void zero_stats_kernel(float* p) {
    int i = threadIdx.x;
    if (i < 320) p[i] = 0.0f;
}

// ---------------------------------------------------------------------------
// Kernel 1: bilinear downsample 1024x1024x64 fp32 -> 512x512x64 bf16.
// Matches the reference quirk: meshgrid('xy') makes the row coordinate vary
// with the SECOND output index (spatial transpose), linspace(0,1024,512)
// puts the last sample out of range -> zero (map_coordinates cval=0).
// ---------------------------------------------------------------------------
__global__ __launch_bounds__(256)
void downsample_kernel(const float* __restrict__ x, __bf16* __restrict__ xds) {
    int idx = blockIdx.x * 256 + threadIdx.x;        // HD*WD*16
    if (idx >= HD * WD * 16) return;
    int c4  = (idx & 15) * 4;
    int pix = idx >> 4;
    int j = pix & (WD - 1);
    int i = pix >> 9;

    const float step = (float)WIN / (float)(WD - 1); // 1024/511
    float r = j * step;   // row coord varies with j (transpose quirk)
    float c = i * step;
    int   r0 = (int)floorf(r), c0 = (int)floorf(c);
    float fr = r - (float)r0, fc = c - (float)c0;

    float a0 = 0.f, a1 = 0.f, a2 = 0.f, a3 = 0.f;
    #pragma unroll
    for (int dr = 0; dr < 2; ++dr) {
        #pragma unroll
        for (int dc = 0; dc < 2; ++dc) {
            int rr = r0 + dr, cc = c0 + dc;
            float w = (dr ? fr : 1.0f - fr) * (dc ? fc : 1.0f - fc);
            if ((unsigned)rr < (unsigned)HIN && (unsigned)cc < (unsigned)WIN) {
                const float4 v = *(const float4*)(x + ((size_t)(rr * WIN + cc) * CIN + c4));
                a0 += w * v.x; a1 += w * v.y; a2 += w * v.z; a3 += w * v.w;
            }
        }
    }
    v4bf o;
    o[0] = (__bf16)a0; o[1] = (__bf16)a1; o[2] = (__bf16)a2; o[3] = (__bf16)a3;
    *(v4bf*)(xds + ((size_t)pix * CIN + c4)) = o;
}

// ---------------------------------------------------------------------------
// Kernel 2: fold the 1x1 conv into the 3x3 center tap, transpose weights to
// [tap][cout][cin] bf16 so B fragments are contiguous-K reads.
// ---------------------------------------------------------------------------
__global__ __launch_bounds__(256)
void prep_weights_kernel(const float* __restrict__ W1, const float* __restrict__ W2,
                         __bf16* __restrict__ wt) {
    int idx = blockIdx.x * 256 + threadIdx.x;        // 9*128*64
    if (idx >= 9 * COUT * CIN) return;
    int ci = idx & (CIN - 1);
    int co = (idx >> 6) & (COUT - 1);
    int t  = idx >> 13;                              // tap = ky*3+kx
    float w = W1[((size_t)t * CIN + ci) * COUT + co];
    if (t == 4) w += W2[(size_t)ci * COUT + co];     // fold 1x1 conv
    wt[idx] = (__bf16)w;                             // wt[t][co][ci]
}

// ---------------------------------------------------------------------------
// Kernel 3: 3x3 conv as implicit GEMM with V_WMMA_F32_16X16X32_BF16.
// Workgroup: 64 pixels (one row strip) x 32 couts; 8 waves, each wave owns a
// 16-pixel x 16-cout C tile and runs 9 taps x 2 K-chunks = 18 WMMAs (K=576).
// Staging uses the gfx1250 async global->LDS engine when available.
// Fragment layouts per CDNA5 ISA §7.12.2 (wave32):
//   A 16x32 bf16 : lane m=l&15; K elems = {klo..klo+7} U {klo+16..klo+23},
//                  klo = (l>=16)*8  -> two 16B LDS loads
//   B 32x16 bf16 : lane n=l&15; K elems = same as A (klo..klo+7, klo+16..klo+23)
//   C 16x16 f32  : VGPR r -> M = r + (l>=16)*8, N = l&15
// Writes the pre-GroupNorm fp32 result straight into d_out.
// ---------------------------------------------------------------------------
__global__ __launch_bounds__(256)
void conv_wmma_kernel(const __bf16* __restrict__ xds,
                      const __bf16* __restrict__ wt,
                      const float* __restrict__ b1,
                      const float* __restrict__ b2,
                      float* __restrict__ y) {
    __shared__ alignas(16) __bf16 in_tile[3][66][CIN];  // 25344 B (halo strip)
    __shared__ alignas(16) __bf16 w_tile[9][32][CIN];   // 36864 B

    const int xb       = blockIdx.x * 64;     // pixel-x base
    const int row      = blockIdx.y;          // output row
    const int coutBase = blockIdx.z * 32;     // cout base
    const int tid      = threadIdx.x;

    // Warm L2 for the workgroup that will handle the next row of this strip.
    if (tid == 0 && row + 1 < HD)
        __builtin_prefetch(xds + (((size_t)(row + 1) * WD + xb) * CIN), 0, 1);

    // ---- stage input strip: rows row-1..row+1, cols xb-1..xb+64, zero pad ----
    for (int chunk = tid; chunk < 3 * 66 * 8; chunk += 256) {
        int dy  = chunk / (66 * 8);
        int r   = chunk % (66 * 8);
        int col = r >> 3;             // 0..65
        int ch8 = (r & 7) * 8;        // channel octet
        int gy = row - 1 + dy;
        int gx = xb - 1 + col;
        __bf16* dst = &in_tile[dy][col][ch8];
        if ((unsigned)gy < (unsigned)HD && (unsigned)gx < (unsigned)WD) {
            g2l_16B(xds + (((size_t)gy * WD + gx) * CIN + ch8), dst);
        } else {
            *(v8bf*)dst = (v8bf){};   // halo zero-fill (cval=0)
        }
    }
    // ---- stage this block's 32-cout weight slab (always in-bounds) ----
    for (int chunk = tid; chunk < 9 * 32 * 8; chunk += 256) {
        int t   = chunk / (32 * 8);
        int r   = chunk % (32 * 8);
        int co  = r >> 3;
        int ch8 = (r & 7) * 8;
        g2l_16B(wt + (((size_t)t * COUT + coutBase + co) * CIN + ch8),
                &w_tile[t][co][ch8]);
    }
    wait_async0();          // drain ASYNCcnt before the barrier
    __syncthreads();

    const int wave = tid >> 5;
    const int lane = tid & 31;
    const int psub = wave & 3;        // 16-pixel sub tile
    const int csub = wave >> 2;       // 16-cout sub tile
    const int ln   = lane & 15;
    const int hi   = lane >> 4;
    const int klo  = hi * 8;

    const int ch = coutBase + csub * 16 + ln;     // this lane's output channel
    const float bias = b1[ch] + b2[ch];
    v8f acc = {};
    #pragma unroll
    for (int e = 0; e < 8; ++e) acc[e] = bias;

    #pragma unroll
    for (int dy = 0; dy < 3; ++dy) {
        #pragma unroll
        for (int dx = 0; dx < 3; ++dx) {
            const int t = dy * 3 + dx;
            const __bf16* arow = &in_tile[dy][psub * 16 + ln + dx][0];
            const __bf16* brow = &w_tile[t][csub * 16 + ln][0];
            #pragma unroll
            for (int kc = 0; kc < CIN; kc += 32) {
                union { v16bf v; v8bf h[2]; } A, B;
                A.h[0] = *(const v8bf*)(arow + kc + klo);
                A.h[1] = *(const v8bf*)(arow + kc + klo + 16);
                B.h[0] = *(const v8bf*)(brow + kc + klo);
                B.h[1] = *(const v8bf*)(brow + kc + klo + 16);
                acc = __builtin_amdgcn_wmma_f32_16x16x32_bf16(
                    /*neg_a=*/false, A.v, /*neg_b=*/false, B.v,
                    /*c_mod=*/(short)0, acc, /*reuse_a=*/false, /*reuse_b=*/false);
            }
        }
    }

    // ---- write C tile: element r -> pixel (r + hi*8), channel ch ----
    const size_t pixBase = (size_t)row * WD + xb + psub * 16;
    float* dst = y + (pixBase + (size_t)hi * 8) * COUT + ch;
    #pragma unroll
    for (int r = 0; r < 8; ++r)
        dst[(size_t)r * COUT] = acc[r];
}

// ---------------------------------------------------------------------------
// Kernel 4: per-channel sum / sum-of-squares over all pixels (atomics)
// ---------------------------------------------------------------------------
__global__ __launch_bounds__(256)
void stats_kernel(const float* __restrict__ y, float* sums, float* sumsq) {
    __shared__ float sred[256], s2red[256];
    const int c    = threadIdx.x & (COUT - 1);
    const int slot = threadIdx.x >> 7;           // 0..1
    float s = 0.f, s2 = 0.f;
    for (int pix = blockIdx.x * 2 + slot; pix < HD * WD; pix += gridDim.x * 2) {
        float v = y[(size_t)pix * COUT + c];
        s += v; s2 += v * v;
    }
    sred[threadIdx.x] = s; s2red[threadIdx.x] = s2;
    __syncthreads();
    if (slot == 0) {
        s  += sred[128 + c];
        s2 += s2red[128 + c];
        atomicAdd(&sums[c], s);
        atomicAdd(&sumsq[c], s2);
    }
}

// ---------------------------------------------------------------------------
// Kernel 5: fold channel stats into 32 group (mean, rsqrt(var+eps))
// ---------------------------------------------------------------------------
__global__ void finalize_stats_kernel(const float* sums, const float* sumsq,
                                      float* gmean, float* grstd) {
    int g = threadIdx.x;
    if (g >= NGRP) return;
    float s = 0.f, s2 = 0.f;
    #pragma unroll
    for (int k = 0; k < 4; ++k) { s += sums[g * 4 + k]; s2 += sumsq[g * 4 + k]; }
    const float invN = 1.0f / ((float)HD * (float)WD * 4.0f);
    float m   = s * invN;
    float var = s2 * invN - m * m;
    gmean[g] = m;
    grstd[g] = rsqrtf(var + GEPS);
}

// ---------------------------------------------------------------------------
// Kernel 6: in-place GroupNorm affine + tanh-GELU. One float4 == exactly one
// group's 4 channels, so each thread needs a single (mean, rstd) pair.
// ---------------------------------------------------------------------------
__device__ __forceinline__ float gelu_tanh(float t) {
    const float k0 = 0.7978845608028654f;   // sqrt(2/pi)
    const float k1 = 0.044715f;
    return 0.5f * t * (1.0f + tanhf(k0 * (t + k1 * t * t * t)));
}

__global__ __launch_bounds__(256)
void gn_gelu_kernel(float* __restrict__ y,
                    const float* __restrict__ gmean, const float* __restrict__ grstd,
                    const float* __restrict__ scale, const float* __restrict__ bias) {
    size_t idx = (size_t)blockIdx.x * 256 + threadIdx.x;   // HD*WD*32 float4s
    if (idx >= (size_t)HD * WD * (COUT / 4)) return;
    const int g = (int)(idx & (NGRP - 1));                 // group index
    float4 v = ((float4*)y)[idx];
    const float m = gmean[g], rs = grstd[g];
    const int c = g * 4;
    v.x = gelu_tanh((v.x - m) * rs * scale[c + 0] + bias[c + 0]);
    v.y = gelu_tanh((v.y - m) * rs * scale[c + 1] + bias[c + 1]);
    v.z = gelu_tanh((v.z - m) * rs * scale[c + 2] + bias[c + 2]);
    v.w = gelu_tanh((v.w - m) * rs * scale[c + 3] + bias[c + 3]);
    ((float4*)y)[idx] = v;
}

// ---------------------------------------------------------------------------
// Host launcher
// ---------------------------------------------------------------------------
extern "C" void kernel_launch(void* const* d_in, const int* in_sizes, int n_in,
                              void* d_out, int out_size, void* d_ws, size_t ws_size,
                              hipStream_t stream) {
    (void)in_sizes; (void)n_in; (void)out_size; (void)ws_size;
    const float* x      = (const float*)d_in[0];
    const float* W1     = (const float*)d_in[1];
    const float* b1     = (const float*)d_in[2];
    const float* W2     = (const float*)d_in[3];
    const float* b2     = (const float*)d_in[4];
    const float* gscale = (const float*)d_in[5];
    const float* gbias  = (const float*)d_in[6];
    float* y = (float*)d_out;

    char* ws = (char*)d_ws;
    __bf16* xds  = (__bf16*)ws;                                     // 33,554,432 B
    __bf16* wtw  = (__bf16*)(ws + (size_t)HD * WD * CIN * 2);       //    147,456 B
    float*  stat = (float*)(ws + (size_t)HD * WD * CIN * 2 + 9 * COUT * CIN * 2);
    float* sums  = stat;
    float* sumsq = stat + 128;
    float* gmean = stat + 256;
    float* grstd = stat + 288;

    zero_stats_kernel<<<1, 320, 0, stream>>>(stat);
    downsample_kernel<<<(HD * WD * 16) / 256, 256, 0, stream>>>(x, xds);
    prep_weights_kernel<<<(9 * COUT * CIN + 255) / 256, 256, 0, stream>>>(W1, W2, wtw);

    dim3 cgrid(WD / 64, HD, COUT / 32);          // 8 x 512 x 4
    conv_wmma_kernel<<<cgrid, 256, 0, stream>>>(xds, wtw, b1, b2, y);

    stats_kernel<<<512, 256, 0, stream>>>(y, sums, sumsq);
    finalize_stats_kernel<<<1, 32, 0, stream>>>(sums, sumsq, gmean, grstd);
    gn_gelu_kernel<<<(HD * WD * (COUT / 4)) / 256, 256, 0, stream>>>(y, gmean, grstd, gscale, gbias);
}
